// AxialAttention_59554016526730
// MI455X (gfx1250) — compile-verified, matrix operand order checked by reading, not run on hardware
//
#include <hip/hip_runtime.h>
#include <math.h>

// ---------------------------------------------------------------------------
// Problem constants (from reference)
// ---------------------------------------------------------------------------
#define NB      8      // N_BATCH
#define CIN     512    // IN_PLANES
#define HH      64     // K (sequence length along axial dim)
#define WW      64     // W_DIM
#define BT      512    // B = NB * WW
#define GROUPS  8
#define GP      64
#define OC2     1024   // OUT_PLANES*2
#define EPS     1e-5f

typedef __attribute__((ext_vector_type(16))) __bf16 v16bf;
typedef __attribute__((ext_vector_type(8)))  float  v8f;

// ---------------------------------------------------------------------------
// WMMA helpers (CDNA5 bf16 16x16x32, f32 accumulate)
// ---------------------------------------------------------------------------
__device__ __forceinline__ v8f wmma_bf16(v16bf a, v16bf b, v8f c) {
  return __builtin_amdgcn_wmma_f32_16x16x32_bf16(false, a, false, b,
                                                 (short)0, c, false, false);
}

// A fragment: 16x32 tile of row-major A[m][k]; lane l<16 holds row m=l,
// K in {0..7,16..23}; lane>=16 holds row m=l-16, K in {8..15,24..31}.
// Pairs (kb,kb+1) are contiguous -> packed dword loads.
__device__ __forceinline__ v16bf load_a_bf(const __bf16* base, int ld,
                                           int m0, int k0) {
  int lane = threadIdx.x & 31;
  int m    = m0 + (lane & 15);
  int grp  = lane >> 4;
  union { v16bf v; unsigned u[8]; } r;
  const __bf16* row = base + (size_t)m * ld + k0;
#pragma unroll
  for (int t = 0; t < 8; ++t) {
    int kb = (t < 4 ? 2 * t : 16 + 2 * (t - 4)) + 8 * grp;
    r.u[t] = *(const unsigned*)(row + kb);
  }
  return r.v;
}

// B fragment: 32x16 tile of row-major B[k][n] (mirror of A with n in place
// of m; K pattern identical).
__device__ __forceinline__ v16bf load_b_bf(const __bf16* base, int ld,
                                           int n0, int k0) {
  int lane = threadIdx.x & 31;
  int n    = n0 + (lane & 15);
  int grp  = lane >> 4;
  v16bf f;
#pragma unroll
  for (int t = 0; t < 8; ++t) {
    int kb = (t < 4 ? 2 * t : 16 + 2 * (t - 4)) + 8 * grp + k0;
    f[2 * t]     = base[(size_t)kb * ld + n];
    f[2 * t + 1] = base[(size_t)(kb + 1) * ld + n];
  }
  return f;
}

// B fragment from f32 LDS matrix read TRANSPOSED: B[k][n] = S[n][k]
// (used for sim^T with S = sim[i][j], ld fixed = 64)
__device__ __forceinline__ v16bf load_b_f32T(const float* S, int n0, int k0) {
  int lane = threadIdx.x & 31;
  int n    = n0 + (lane & 15);
  int grp  = lane >> 4;
  v16bf f;
#pragma unroll
  for (int t = 0; t < 8; ++t) {
    int kb = (t < 4 ? 2 * t : 16 + 2 * (t - 4)) + 8 * grp + k0;
    f[2 * t]     = (__bf16)S[n * 64 + kb];
    f[2 * t + 1] = (__bf16)S[n * 64 + kb + 1];
  }
  return f;
}

__device__ __forceinline__ float lane_reduce_add(float v) {
#pragma unroll
  for (int off = 16; off > 0; off >>= 1) v += __shfl_xor(v, off, 32);
  return v;
}

// ---------------------------------------------------------------------------
// Prep kernels
// ---------------------------------------------------------------------------
__global__ void f32_to_bf16_k(const float* __restrict__ src,
                              __bf16* __restrict__ dst, int n) {
  int i = blockIdx.x * 256 + threadIdx.x;
  if (i < n) dst[i] = (__bf16)src[i];
}

// all_emb[c][i][j] = relative[c][i-j+63] + pos_map[c][i][j], c in [0,128)
__global__ void build_emb_k(const float* __restrict__ relative,
                            const float* __restrict__ pos_map,
                            float* __restrict__ emb) {
  int c = blockIdx.x;
  for (int idx = threadIdx.x; idx < HH * HH; idx += 256) {
    int i = idx >> 6, j = idx & 63;
    emb[(size_t)c * HH * HH + idx] =
        relative[(size_t)c * (2 * HH - 1) + (i - j + HH - 1)] +
        pos_map[(size_t)c * HH * HH + idx];
  }
}

// xp[b][c][h] = (bf16) x[n][c][h][w],  b = n*64 + w. LDS-tiled transpose.
__global__ void build_xp_k(const float* __restrict__ x,
                           __bf16* __restrict__ xp) {
  int n = blockIdx.z, c = blockIdx.y;
  int th = (blockIdx.x >> 2) * 16, tw = (blockIdx.x & 3) * 16;
  __shared__ float t[16][17];
  int h = th + threadIdx.y, w = tw + threadIdx.x;
  t[threadIdx.y][threadIdx.x] =
      x[(((size_t)n * CIN + c) * HH + h) * WW + w];
  __syncthreads();
  int w2 = tw + threadIdx.y, h2 = th + threadIdx.x;
  xp[(((size_t)n * WW + w2) * CIN + c) * HH + h2] =
      (__bf16)t[threadIdx.x][threadIdx.y];
}

// ---------------------------------------------------------------------------
// QKV GEMM: qkv[b][o][h] = sum_c qkv_w[o][c] * xp[b][c][h]
// grid (512, 4), 128 threads = 4 waves; each wave -> 64(o) x 64(h) tile.
// ---------------------------------------------------------------------------
__global__ __launch_bounds__(128) void qkv_gemm_k(
    const __bf16* __restrict__ wq, const __bf16* __restrict__ xp,
    float* __restrict__ qkv) {
  int b    = blockIdx.x;
  int wave = threadIdx.x >> 5;
  int o0   = blockIdx.y * 256 + wave * 64;
  const __bf16* Ab = wq + (size_t)o0 * CIN;
  const __bf16* Bb = xp + (size_t)b * CIN * HH;

  v8f zero = {0.f, 0.f, 0.f, 0.f, 0.f, 0.f, 0.f, 0.f};
  v8f acc[4][4];
#pragma unroll
  for (int i = 0; i < 4; ++i)
#pragma unroll
    for (int j = 0; j < 4; ++j) acc[i][j] = zero;

  for (int k0 = 0; k0 < CIN; k0 += 32) {
    v16bf af[4], bf_[4];
#pragma unroll
    for (int t = 0; t < 4; ++t) af[t]  = load_a_bf(Ab, CIN, t * 16, k0);
#pragma unroll
    for (int t = 0; t < 4; ++t) bf_[t] = load_b_bf(Bb, HH, t * 16, k0);
#pragma unroll
    for (int ti = 0; ti < 4; ++ti)
#pragma unroll
      for (int tj = 0; tj < 4; ++tj)
        acc[ti][tj] = wmma_bf16(af[ti], bf_[tj], acc[ti][tj]);
  }

  int lane = threadIdx.x & 31, n = lane & 15, grp = lane >> 4;
#pragma unroll
  for (int ti = 0; ti < 4; ++ti)
#pragma unroll
    for (int tj = 0; tj < 4; ++tj)
#pragma unroll
      for (int r = 0; r < 8; ++r) {
        int o = o0 + ti * 16 + r + 8 * grp;
        int h = tj * 16 + n;
        qkv[((size_t)b * OC2 + o) * HH + h] = acc[ti][tj][r];
      }
}

// ---------------------------------------------------------------------------
// Per-channel mean/rstd over (b, h) for a (B, C, 64) tensor. grid = C.
// ---------------------------------------------------------------------------
__global__ __launch_bounds__(256) void channel_stats_k(
    const float* __restrict__ src, float* __restrict__ stat, int C) {
  int o = blockIdx.x, tid = threadIdx.x;
  float s = 0.f, s2 = 0.f;
  for (int idx = tid; idx < BT * HH; idx += 256) {
    int b = idx >> 6, h = idx & 63;
    float v = src[((size_t)b * C + o) * HH + h];
    s += v; s2 += v * v;
  }
  __shared__ float r0[256], r1[256];
  r0[tid] = s; r1[tid] = s2;
  __syncthreads();
  for (int off = 128; off > 0; off >>= 1) {
    if (tid < off) { r0[tid] += r0[tid + off]; r1[tid] += r1[tid + off]; }
    __syncthreads();
  }
  if (tid == 0) {
    float invc = 1.f / (float)(BT * HH);
    float mean = r0[0] * invc;
    float var  = r1[0] * invc - mean * mean;
    stat[2 * o]     = mean;
    stat[2 * o + 1] = rsqrtf(var + EPS);
  }
}

// ---------------------------------------------------------------------------
// Shared: load BN-normalized q (transposed), k, (v) into LDS as bf16
// ---------------------------------------------------------------------------
__device__ __forceinline__ void load_qk_lds(
    const float* qkv, const float* qstat, const float* gq, const float* bq,
    int b, int g, __bf16* qT, __bf16* kn) {
  int tid = threadIdx.x;
  size_t qbase = ((size_t)b * OC2 + g * 128) * HH;
  for (int it = 0; it < 64; ++it) {            // q: 32x64
    int idx = it * 32 + tid, c = idx >> 6, i = idx & 63;
    int o = g * 128 + c;
    float v  = qkv[qbase + (size_t)c * HH + i];
    float nv = (v - qstat[2 * o]) * qstat[2 * o + 1] * gq[o] + bq[o];
    qT[i * 32 + c] = (__bf16)nv;               // A layout: qT[i][c]
  }
  for (int it = 0; it < 64; ++it) {            // k: 32x64
    int idx = it * 32 + tid, c = idx >> 6, j = idx & 63;
    int o = g * 128 + 32 + c;
    float v  = qkv[qbase + (size_t)(32 + c) * HH + j];
    float nv = (v - qstat[2 * o]) * qstat[2 * o + 1] * gq[o] + bq[o];
    kn[c * 64 + j] = (__bf16)nv;               // B layout: kn[c][j]
  }
}

// ---------------------------------------------------------------------------
// Pass A: recompute qk/qr/kr, emit per-(b,g) partial sums for sim-BN stats.
// grid (512, 8), 32 threads (one wave).
// ---------------------------------------------------------------------------
__global__ __launch_bounds__(32) void attn_stats_k(
    const float* __restrict__ qkv, const float* __restrict__ qstat,
    const float* __restrict__ gq, const float* __restrict__ bq,
    const float* __restrict__ emb, float* __restrict__ part) {
  int b = blockIdx.x, g = blockIdx.y, lane = threadIdx.x;
  __shared__ __bf16 qT[64 * 32];
  __shared__ __bf16 kn[32 * 64];
  load_qk_lds(qkv, qstat, gq, bq, b, g, qT, kn);
  __syncthreads();

  // qk = q^T k  via WMMA
  v8f zero = {0.f, 0.f, 0.f, 0.f, 0.f, 0.f, 0.f, 0.f};
  v8f acc[4][4];
#pragma unroll
  for (int i = 0; i < 4; ++i)
#pragma unroll
    for (int j = 0; j < 4; ++j) acc[i][j] = zero;
  v16bf bfr[4];
#pragma unroll
  for (int tj = 0; tj < 4; ++tj) bfr[tj] = load_b_bf(kn, 64, tj * 16, 0);
#pragma unroll
  for (int ti = 0; ti < 4; ++ti) {
    v16bf a = load_a_bf(qT, 32, ti * 16, 0);
#pragma unroll
    for (int tj = 0; tj < 4; ++tj)
      acc[ti][tj] = wmma_bf16(a, bfr[tj], acc[ti][tj]);
  }
  float sqk = 0.f, s2qk = 0.f;
#pragma unroll
  for (int ti = 0; ti < 4; ++ti)
#pragma unroll
    for (int tj = 0; tj < 4; ++tj)
#pragma unroll
      for (int r = 0; r < 8; ++r) {
        float v = acc[ti][tj][r];
        sqk += v; s2qk += v * v;
      }

  // qr[i][j] = sum_c q[c][i] * emb_q[c][i][j]   (lane = j)
  const float* embq = emb;
  float sqr = 0.f, s2qr = 0.f;
  for (int i = 0; i < 64; ++i)
    for (int jh = 0; jh < 2; ++jh) {
      int j = lane + jh * 32;
      float a = 0.f;
      for (int c = 0; c < 32; ++c)
        a += (float)qT[i * 32 + c] * embq[((size_t)c * 64 + i) * 64 + j];
      sqr += a; s2qr += a * a;
    }

  // kr[i][j] = sum_c k[c][j] * emb_k[c][j][i]   (lane = i)
  const float* embk = emb + (size_t)32 * 4096;
  float skr = 0.f, s2kr = 0.f;
  for (int j = 0; j < 64; ++j)
    for (int ih = 0; ih < 2; ++ih) {
      int i = lane + ih * 32;
      float a = 0.f;
      for (int c = 0; c < 32; ++c)
        a += (float)kn[c * 64 + j] * embk[((size_t)c * 64 + j) * 64 + i];
      skr += a; s2kr += a * a;
    }

  sqk = lane_reduce_add(sqk);  s2qk = lane_reduce_add(s2qk);
  sqr = lane_reduce_add(sqr);  s2qr = lane_reduce_add(s2qr);
  skr = lane_reduce_add(skr);  s2kr = lane_reduce_add(s2kr);
  if (lane == 0) {
    part[((size_t)(0 * 8 + g) * BT + b) * 2]     = sqk;
    part[((size_t)(0 * 8 + g) * BT + b) * 2 + 1] = s2qk;
    part[((size_t)(8 + g) * BT + b) * 2]         = sqr;
    part[((size_t)(8 + g) * BT + b) * 2 + 1]     = s2qr;
    part[((size_t)(16 + g) * BT + b) * 2]        = skr;
    part[((size_t)(16 + g) * BT + b) * 2 + 1]    = s2kr;
  }
}

__global__ __launch_bounds__(256) void sim_reduce_k(
    const float* __restrict__ part, float* __restrict__ stat) {
  int ch = blockIdx.x, tid = threadIdx.x;
  float s = 0.f, s2 = 0.f;
  for (int bidx = tid; bidx < BT; bidx += 256) {
    s  += part[((size_t)ch * BT + bidx) * 2];
    s2 += part[((size_t)ch * BT + bidx) * 2 + 1];
  }
  __shared__ float r0[256], r1[256];
  r0[tid] = s; r1[tid] = s2;
  __syncthreads();
  for (int off = 128; off > 0; off >>= 1) {
    if (tid < off) { r0[tid] += r0[tid + off]; r1[tid] += r1[tid + off]; }
    __syncthreads();
  }
  if (tid == 0) {
    float invc = 1.f / (float)((size_t)BT * HH * HH);
    float mean = r0[0] * invc;
    float var  = r1[0] * invc - mean * mean;
    stat[2 * ch]     = mean;
    stat[2 * ch + 1] = rsqrtf(var + EPS);
  }
}

// ---------------------------------------------------------------------------
// Pass B: full attention per (b,g). grid (512, 8), 32 threads (one wave).
// ---------------------------------------------------------------------------
__global__ __launch_bounds__(32) void attn_main_k(
    const float* __restrict__ qkv, const float* __restrict__ qstat,
    const float* __restrict__ gq, const float* __restrict__ bq,
    const float* __restrict__ emb, const float* __restrict__ sstat,
    const float* __restrict__ gs, const float* __restrict__ bs,
    const __bf16* __restrict__ wbf, float* __restrict__ out_ws) {
  int b = blockIdx.x, g = blockIdx.y, lane = threadIdx.x;
  __shared__ __bf16 qT[64 * 32];
  __shared__ __bf16 kn[32 * 64];
  __shared__ __bf16 vn[64 * 64];
  __shared__ float  sim[64 * 64];
  __shared__ __bf16 svp[64 * 64];
  __shared__ __bf16 svep[64 * 64];

  load_qk_lds(qkv, qstat, gq, bq, b, g, qT, kn);
  {
    size_t qbase = ((size_t)b * OC2 + g * 128) * HH;
    for (int it = 0; it < 128; ++it) {         // v: 64x64
      int idx = it * 32 + lane, c = idx >> 6, j = idx & 63;
      int o = g * 128 + 64 + c;
      float v  = qkv[qbase + (size_t)(64 + c) * HH + j];
      float nv = (v - qstat[2 * o]) * qstat[2 * o + 1] * gq[o] + bq[o];
      vn[c * 64 + j] = (__bf16)nv;
    }
  }
  __syncthreads();

  v8f zero = {0.f, 0.f, 0.f, 0.f, 0.f, 0.f, 0.f, 0.f};
  int n = lane & 15, grp = lane >> 4;

  // ---- qk via WMMA, normalize into sim ----
  {
    v8f acc[4][4];
#pragma unroll
    for (int i = 0; i < 4; ++i)
#pragma unroll
      for (int j = 0; j < 4; ++j) acc[i][j] = zero;
    v16bf bfr[4];
#pragma unroll
    for (int tj = 0; tj < 4; ++tj) bfr[tj] = load_b_bf(kn, 64, tj * 16, 0);
#pragma unroll
    for (int ti = 0; ti < 4; ++ti) {
      v16bf a = load_a_bf(qT, 32, ti * 16, 0);
#pragma unroll
      for (int tj = 0; tj < 4; ++tj)
        acc[ti][tj] = wmma_bf16(a, bfr[tj], acc[ti][tj]);
    }
    float m = sstat[2 * g], rr = sstat[2 * g + 1];
    float gg = gs[g], bb = bs[g];
#pragma unroll
    for (int ti = 0; ti < 4; ++ti)
#pragma unroll
      for (int tj = 0; tj < 4; ++tj)
#pragma unroll
        for (int r = 0; r < 8; ++r) {
          int i = ti * 16 + r + 8 * grp, j = tj * 16 + n;
          sim[i * 64 + j] = (acc[ti][tj][r] - m) * rr * gg + bb;
        }
  }
  __syncthreads();

  // ---- qr (lane = j) ----
  {
    float m = sstat[2 * (8 + g)], rr = sstat[2 * (8 + g) + 1];
    float gg = gs[8 + g], bb = bs[8 + g];
    const float* embq = emb;
    for (int i = 0; i < 64; ++i)
      for (int jh = 0; jh < 2; ++jh) {
        int j = lane + jh * 32;
        float a = 0.f;
        for (int c = 0; c < 32; ++c)
          a += (float)qT[i * 32 + c] * embq[((size_t)c * 64 + i) * 64 + j];
        sim[i * 64 + j] += (a - m) * rr * gg + bb;
      }
  }
  __syncthreads();

  // ---- kr (lane = i) ----
  {
    float m = sstat[2 * (16 + g)], rr = sstat[2 * (16 + g) + 1];
    float gg = gs[16 + g], bb = bs[16 + g];
    const float* embk = emb + (size_t)32 * 4096;
    for (int j = 0; j < 64; ++j)
      for (int ih = 0; ih < 2; ++ih) {
        int i = lane + ih * 32;
        float a = 0.f;
        for (int c = 0; c < 32; ++c)
          a += (float)kn[c * 64 + j] * embk[((size_t)c * 64 + j) * 64 + i];
        sim[i * 64 + j] += (a - m) * rr * gg + bb;
      }
  }
  __syncthreads();

  // ---- softmax over j ----
  for (int i = 0; i < 64; ++i) {
    float v0 = sim[i * 64 + lane], v1 = sim[i * 64 + lane + 32];
    float mx = fmaxf(v0, v1);
#pragma unroll
    for (int off = 16; off > 0; off >>= 1)
      mx = fmaxf(mx, __shfl_xor(mx, off, 32));
    float e0 = __expf(v0 - mx), e1 = __expf(v1 - mx);
    float s = e0 + e1;
#pragma unroll
    for (int off = 16; off > 0; off >>= 1) s += __shfl_xor(s, off, 32);
    float inv = 1.f / s;
    sim[i * 64 + lane]      = e0 * inv;
    sim[i * 64 + lane + 32] = e1 * inv;
  }
  __syncthreads();

  // ---- sv_pre[c][i] = sum_j v[c][j] * sim[i][j]  (WMMA, B = sim^T) ----
  {
    v8f acc[4][4];
#pragma unroll
    for (int i = 0; i < 4; ++i)
#pragma unroll
      for (int j = 0; j < 4; ++j) acc[i][j] = zero;
    for (int kk = 0; kk < 64; kk += 32) {
      v16bf bfr[4];
#pragma unroll
      for (int ti = 0; ti < 4; ++ti) bfr[ti] = load_b_f32T(sim, ti * 16, kk);
#pragma unroll
      for (int tc = 0; tc < 4; ++tc) {
        v16bf a = load_a_bf(vn, 64, tc * 16, kk);
#pragma unroll
        for (int ti = 0; ti < 4; ++ti)
          acc[tc][ti] = wmma_bf16(a, bfr[ti], acc[tc][ti]);
      }
    }
#pragma unroll
    for (int tc = 0; tc < 4; ++tc)
#pragma unroll
      for (int ti = 0; ti < 4; ++ti)
#pragma unroll
        for (int r = 0; r < 8; ++r) {
          int c = tc * 16 + r + 8 * grp, i = ti * 16 + n;
          svp[c * 64 + i] = (__bf16)acc[tc][ti][r];
        }
  }
  __syncthreads();

  // ---- sve_pre[c][i] = sum_j sim[i][j] * emb_v[c][i][j]  (lane = i) ----
  {
    const float* embv = emb + (size_t)64 * 4096;
    for (int c = 0; c < 64; ++c)
      for (int ih = 0; ih < 2; ++ih) {
        int i = lane + ih * 32;
        float a = 0.f;
        for (int j = 0; j < 64; ++j)
          a += sim[i * 64 + j] * embv[((size_t)c * 64 + i) * 64 + j];
        svep[c * 64 + i] = (__bf16)a;
      }
  }
  __syncthreads();

  // ---- sv = sv_pre @ W, sve = sve_pre @ W; interleave into out_ws ----
  size_t obase = ((size_t)b * OC2 + g * 128) * HH;
  for (int srcsel = 0; srcsel < 2; ++srcsel) {
    const __bf16* A = (srcsel == 0) ? svp : svep;
    v8f acc[4][4];
#pragma unroll
    for (int i = 0; i < 4; ++i)
#pragma unroll
      for (int j = 0; j < 4; ++j) acc[i][j] = zero;
    for (int kk = 0; kk < 64; kk += 32) {
      v16bf bw[4];
#pragma unroll
      for (int tm = 0; tm < 4; ++tm) bw[tm] = load_b_bf(wbf, 64, tm * 16, kk);
#pragma unroll
      for (int tc = 0; tc < 4; ++tc) {
        v16bf a = load_a_bf(A, 64, tc * 16, kk);
#pragma unroll
        for (int tm = 0; tm < 4; ++tm)
          acc[tc][tm] = wmma_bf16(a, bw[tm], acc[tc][tm]);
      }
    }
#pragma unroll
    for (int tc = 0; tc < 4; ++tc)
#pragma unroll
      for (int tm = 0; tm < 4; ++tm)
#pragma unroll
        for (int r = 0; r < 8; ++r) {
          int c = tc * 16 + r + 8 * grp, m = tm * 16 + n;
          out_ws[obase + (size_t)(2 * c + srcsel) * HH + m] = acc[tc][tm][r];
        }
  }
}

// ---------------------------------------------------------------------------
// Final: BN-out, sum channel pairs, transpose (h<->w via LDS), residual.
// grid = N*512, block 256.
// ---------------------------------------------------------------------------
__global__ __launch_bounds__(256) void final_k(
    const float* __restrict__ x, const float* __restrict__ out_ws,
    const float* __restrict__ ostat, const float* __restrict__ go,
    const float* __restrict__ bo, const float* __restrict__ gamma_p,
    float* __restrict__ y) {
  int nb = blockIdx.x >> 9;          // n
  int u  = blockIdx.x & 511;         // out plane (512)
  __shared__ float t[64 * 65];
  float gamma = gamma_p[0];
  int o0 = 2 * u, o1 = 2 * u + 1;
  float m0 = ostat[2 * o0], r0 = ostat[2 * o0 + 1];
  float m1 = ostat[2 * o1], r1 = ostat[2 * o1 + 1];
  float g0 = go[o0], b0 = bo[o0], g1 = go[o1], b1 = bo[o1];

  for (int idx = threadIdx.x; idx < 4096; idx += 256) {
    int w = idx >> 6, h = idx & 63;
    size_t base = (size_t)(nb * 64 + w) * OC2 * HH;
    float v0 = out_ws[base + (size_t)o0 * HH + h];
    float v1 = out_ws[base + (size_t)o1 * HH + h];
    t[w * 65 + h] = (v0 - m0) * r0 * g0 + b0 + (v1 - m1) * r1 * g1 + b1;
  }
  __syncthreads();
  for (int idx = threadIdx.x; idx < 4096; idx += 256) {
    int h = idx >> 6, w = idx & 63;
    size_t xi = (((size_t)nb * CIN + u) * HH + h) * WW + w;
    y[xi] = x[xi] + gamma * t[w * 65 + h];
  }
}

// ---------------------------------------------------------------------------
// Launch
// ---------------------------------------------------------------------------
extern "C" void kernel_launch(void* const* d_in, const int* in_sizes, int n_in,
                              void* d_out, int out_size, void* d_ws,
                              size_t ws_size, hipStream_t stream) {
  const float* x        = (const float*)d_in[0];
  const float* qkv_w    = (const float*)d_in[1];
  const float* bn_qkv_g = (const float*)d_in[2];
  const float* bn_qkv_b = (const float*)d_in[3];
  const float* bn_sim_g = (const float*)d_in[4];
  const float* bn_sim_b = (const float*)d_in[5];
  const float* bn_out_g = (const float*)d_in[6];
  const float* bn_out_b = (const float*)d_in[7];
  const float* weight   = (const float*)d_in[8];
  const float* relative = (const float*)d_in[9];
  const float* gamma    = (const float*)d_in[10];
  const float* pos_map  = (const float*)d_in[11];
  float* y = (float*)d_out;

  char* w = (char*)d_ws;
  auto alloc = [&](size_t bytes) {
    char* p = w;
    w += (bytes + 255) & ~(size_t)255;
    return p;
  };
  __bf16* xp       = (__bf16*)alloc((size_t)BT * CIN * HH * 2);
  __bf16* qkvw_bf  = (__bf16*)alloc((size_t)OC2 * CIN * 2);
  __bf16* w_bf     = (__bf16*)alloc((size_t)GP * GP * 2);
  float*  emb      = (float*)alloc((size_t)128 * HH * HH * 4);
  float*  qkv      = (float*)alloc((size_t)BT * OC2 * HH * 4);
  float*  qstat    = (float*)alloc((size_t)OC2 * 2 * 4);
  float*  part     = (float*)alloc((size_t)24 * BT * 2 * 4);
  float*  sstat    = (float*)alloc((size_t)24 * 2 * 4);
  float*  out_ws   = (float*)alloc((size_t)BT * OC2 * HH * 4);
  float*  ostat    = (float*)alloc((size_t)OC2 * 2 * 4);

  // prep
  f32_to_bf16_k<<<(OC2 * CIN + 255) / 256, 256, 0, stream>>>(qkv_w, qkvw_bf,
                                                             OC2 * CIN);
  f32_to_bf16_k<<<(GP * GP + 255) / 256, 256, 0, stream>>>(weight, w_bf,
                                                           GP * GP);
  build_emb_k<<<128, 256, 0, stream>>>(relative, pos_map, emb);
  build_xp_k<<<dim3(16, CIN, NB), dim3(16, 16), 0, stream>>>(x, xp);

  // qkv GEMM + BN stats
  qkv_gemm_k<<<dim3(BT, 4), 128, 0, stream>>>(qkvw_bf, xp, qkv);
  channel_stats_k<<<OC2, 256, 0, stream>>>(qkv, qstat, OC2);

  // sim BN stats (recompute pass) + reduce
  attn_stats_k<<<dim3(BT, GROUPS), 32, 0, stream>>>(qkv, qstat, bn_qkv_g,
                                                    bn_qkv_b, emb, part);
  sim_reduce_k<<<24, 256, 0, stream>>>(part, sstat);

  // main attention
  attn_main_k<<<dim3(BT, GROUPS), 32, 0, stream>>>(
      qkv, qstat, bn_qkv_g, bn_qkv_b, emb, sstat, bn_sim_g, bn_sim_b, w_bf,
      out_ws);

  // out BN stats + final fuse
  channel_stats_k<<<OC2, 256, 0, stream>>>(out_ws, ostat, OC2);
  final_k<<<NB * 512, 256, 0, stream>>>(x, out_ws, ostat, bn_out_g, bn_out_b,
                                        gamma, y);
}
